// SpatialVaryingVisualQuery_49160195670430
// MI455X (gfx1250) — matrix operand order repaired and checked by reading, not compiled, not run on hardware
//
#include <hip/hip_runtime.h>

// ---------------------------------------------------------------------------
// Problem constants (from reference)
// ---------------------------------------------------------------------------
#define BATCH   4
#define HW      64
#define NQ      65536
#define CDIM    256
#define INDIM   266     // 256 feat + 2 rel + 8 geo
#define KPAD    288     // padded to multiple of 32 for WMMA K
#define HID     256
#define MROWS   32      // queries per workgroup (2 row-tiles of 16)

typedef __attribute__((ext_vector_type(16))) _Float16 v16h;
typedef __attribute__((ext_vector_type(8)))  _Float16 v8h;
typedef __attribute__((ext_vector_type(8)))  float    v8f;

// LDS strides (halves); multiples of 8 so per-row bases stay 16B aligned
#define LDX 296
#define LDH 264
#define LDO 260  // f32 stride for final tile

// ---------------------------------------------------------------------------
// Weight pack kernel: fp32 (K x 256) row-major -> f16 B-fragment-major.
// Fragment address: ((ks*16 + nt)*32 + lane)*16 + e
// holds W[ks*32 + (e/8)*16 + (lane/16)*8 + e%8][nt*16 + lane%16]  (zero-padded K rows).
// Matches the CDNA5 16-bit B-matrix VGPR layout for v_wmma_f32_16x16x32_f16.
// ---------------------------------------------------------------------------
__global__ void pack_w_kernel(const float* __restrict__ W, _Float16* __restrict__ dst,
                              int ksteps, int krows)
{
    int t = blockIdx.x * blockDim.x + threadIdx.x;
    int total = ksteps * 16 * 32 * 16;
    if (t >= total) return;
    int e    =  t        & 15;
    int lane = (t >> 4)  & 31;
    int nt   = (t >> 9)  & 15;
    int ks   =  t >> 13;
    int k    = ks * 32 + (e >> 3) * 16 + (lane >> 4) * 8 + (e & 7);
    int col  = nt * 16 + (lane & 15);
    float v  = (k < krows) ? W[(size_t)k * 256 + col] : 0.0f;
    dst[t] = (_Float16)v;
}

// ---------------------------------------------------------------------------
// CDNA5 async global -> LDS copy (ASYNCcnt-tracked), per-lane 16B chunks.
// VDST = LDS byte address VGPR, VADDR = 64-bit global address VGPR pair.
// Generic LDS pointers carry the LDS byte offset in their low 32 bits.
// ---------------------------------------------------------------------------
__device__ __forceinline__ unsigned lds_off_u32(const void* p)
{
    return (unsigned)(unsigned long long)p;
}
__device__ __forceinline__ void async_ld16(unsigned lds_addr, const void* gptr)
{
    asm volatile("global_load_async_to_lds_b128 %0, %1, off"
                 :: "v"(lds_addr), "v"(gptr) : "memory");
}
__device__ __forceinline__ void wait_async0()
{
    asm volatile("s_wait_asynccnt 0x0" ::: "memory");
}

// ---------------------------------------------------------------------------
// A-fragment load from LDS: lane (h,n) -> row M=n, K halves
//   [kbase + h*8, +8) and [kbase + 16 + h*8, +8)  (CDNA5 16-bit A layout)
// ---------------------------------------------------------------------------
__device__ __forceinline__ v16h load_A(const _Float16* ab, int lda, int m, int kofs)
{
    v8h lo = *(const v8h*)(ab + m * lda + kofs);
    v8h hi = *(const v8h*)(ab + m * lda + kofs + 16);
    return __builtin_shufflevector(lo, hi, 0,1,2,3,4,5,6,7,8,9,10,11,12,13,14,15);
}

// One wave computes a 32 x 32 tile: 2 row-tiles x 2 n-tiles. Each B fragment
// feeds two WMMAs and each A fragment feeds two WMMAs (2x arithmetic
// intensity vs a 16-row tile), with 4 independent accumulator chains for
// deep WMMA pipelining.
__device__ __forceinline__ void gemm4(const _Float16* __restrict__ ab, int lda,
                                      const _Float16* __restrict__ wf, int ksteps,
                                      int lane, int nn, int hh, int nt0, int nt1,
                                      v8f& a00, v8f& a01, v8f& a10, v8f& a11)
{
    const v16h* __restrict__ bp = (const v16h*)wf;
    for (int ks = 0; ks < ksteps; ++ks) {
        int kofs = ks * 32 + hh * 8;
        v16h A0 = load_A(ab,            lda, nn, kofs);
        v16h A1 = load_A(ab + 16 * lda, lda, nn, kofs);
        v16h B0 = bp[(ks * 16 + nt0) * 32 + lane];
        v16h B1 = bp[(ks * 16 + nt1) * 32 + lane];
        a00 = __builtin_amdgcn_wmma_f32_16x16x32_f16(false, A0, false, B0,
                                                     (short)0, a00, false, false);
        a01 = __builtin_amdgcn_wmma_f32_16x16x32_f16(false, A0, false, B1,
                                                     (short)0, a01, false, false);
        a10 = __builtin_amdgcn_wmma_f32_16x16x32_f16(false, A1, false, B0,
                                                     (short)0, a10, false, false);
        a11 = __builtin_amdgcn_wmma_f32_16x16x32_f16(false, A1, false, B1,
                                                     (short)0, a11, false, false);
    }
}

// Store 32x32 C/D tiles back to LDS as f16 with bias (+ optional ReLU).
// C layout: VGPR r holds element (M = r + 8*h, N = nt*16 + n); second
// row-tile at M + 16.
__device__ __forceinline__ void store_h4(_Float16* hb, int ldh,
                                         const v8f& a00, const v8f& a01,
                                         const v8f& a10, const v8f& a11,
                                         const float* __restrict__ bias,
                                         int nt0, int nt1, int hh, int nn, bool relu)
{
    float bb0 = bias[nt0 * 16 + nn];
    float bb1 = bias[nt1 * 16 + nn];
#pragma unroll
    for (int r = 0; r < 8; ++r) {
        int row0 = r + 8 * hh;
        int row1 = row0 + 16;
        float v00 = a00[r] + bb0, v01 = a01[r] + bb1;
        float v10 = a10[r] + bb0, v11 = a11[r] + bb1;
        if (relu) {
            v00 = fmaxf(v00, 0.0f); v01 = fmaxf(v01, 0.0f);
            v10 = fmaxf(v10, 0.0f); v11 = fmaxf(v11, 0.0f);
        }
        hb[row0 * ldh + nt0 * 16 + nn] = (_Float16)v00;
        hb[row0 * ldh + nt1 * 16 + nn] = (_Float16)v01;
        hb[row1 * ldh + nt0 * 16 + nn] = (_Float16)v10;
        hb[row1 * ldh + nt1 * 16 + nn] = (_Float16)v11;
    }
}

// ---------------------------------------------------------------------------
// Main kernel: 256 threads (8 wave32), 32 queries per workgroup.
// ---------------------------------------------------------------------------
__global__ __launch_bounds__(256) void svq_main_kernel(
    const float* __restrict__ feat, const float* __restrict__ coord,
    const float* __restrict__ geo,
    const float* __restrict__ b0, const float* __restrict__ b1,
    const float* __restrict__ b2, const float* __restrict__ bout,
    const float* __restrict__ gamma, const float* __restrict__ beta,
    const _Float16* __restrict__ w0f, const _Float16* __restrict__ w1f,
    const _Float16* __restrict__ w2f, const _Float16* __restrict__ w3f,
    float* __restrict__ out)
{
    // sbuf (async staging, corners) and outf (final tile) have disjoint
    // lifetimes -> overlap them to save LDS.
    __shared__ union {
        float sbuf[MROWS * 256];   // raw f32 feature rows
        float outf[MROWS * LDO];   // final f32 tile
    } u;
    __shared__ _Float16 xbuf [MROWS * LDX];
    __shared__ _Float16 hbuf [MROWS * LDH];
    __shared__ _Float16 hbuf2[MROWS * LDH];
    __shared__ float    wcorner[4][MROWS];
    __shared__ float    relbuf [4][MROWS][2];
    __shared__ float    geos[MROWS][8];
    __shared__ int      gidx[4][MROWS];
    __shared__ float    red[2][MROWS][8];
    __shared__ float    mrow[MROWS], irow[MROWS];

    const int tid   = threadIdx.x;
    const int lane  = tid & 31;
    const int wave  = tid >> 5;
    const int hh    = lane >> 4;
    const int nn    = lane & 15;
    const int nt0   = wave * 2;
    const int nt1   = nt0 + 1;
    const int qbase = blockIdx.x * MROWS;
    const int grow  = tid >> 3;   // gather row (0..31), 8 threads per row
    const int gl8   = tid & 7;

    // ---- Phase 0: per-row geometry / gather indices / blend weights ----
    if (tid < MROWS) {
        int row = tid;
        int q   = qbase + row;
        int b   = q >> 16;           // N = 65536
        float c0 = coord[(size_t)q * 2 + 0];
        float c1 = coord[(size_t)q * 2 + 1];
        float g0 = geo  [(size_t)q * 2 + 0];
        float g1 = geo  [(size_t)q * 2 + 1];
        geos[row][0] = g0 + 1.0f; geos[row][1] = g1 + 1.0f;
        geos[row][2] = g0 + 1.0f; geos[row][3] = g1 - 1.0f;
        geos[row][4] = g0 - 1.0f; geos[row][5] = g1 + 1.0f;
        geos[row][6] = g0 - 1.0f; geos[row][7] = g1 - 1.0f;

        const float rr = 1.0f / 64.0f;
        float areas[4];
#pragma unroll
        for (int c = 0; c < 4; ++c) {
            float vx = (c < 2) ? -1.0f : 1.0f;      // outer loop of reference
            float vy = (c & 1) ?  1.0f : -1.0f;     // inner loop
            float cx = fminf(fmaxf(c0 + vx * rr + 1e-6f, -1.0f + 1e-6f), 1.0f - 1e-6f);
            float cy = fminf(fmaxf(c1 + vy * rr + 1e-6f, -1.0f + 1e-6f), 1.0f - 1e-6f);
            int ix = (int)fminf(fmaxf(rintf((cx + 1.0f) * 32.0f - 0.5f), 0.0f), 63.0f);
            int iy = (int)fminf(fmaxf(rintf((cy + 1.0f) * 32.0f - 0.5f), 0.0f), 63.0f);
            float qc0 = -1.0f + rr + 2.0f * rr * (float)iy;  // seq[iy]
            float qc1 = -1.0f + rr + 2.0f * rr * (float)ix;  // seq[ix]
            float r0 = (c0 - qc0) * 64.0f;
            float r1 = (c1 - qc1) * 64.0f;
            relbuf[c][row][0] = r0;
            relbuf[c][row][1] = r1;
            gidx[c][row] = ((b * HW + iy) * HW + ix) * CDIM;
            areas[c] = 1.0f / (fabsf(r0 * r1) + 1e-9f);
        }
        float tot = areas[0] + areas[1] + areas[2] + areas[3];
#pragma unroll
        for (int c = 0; c < 4; ++c) wcorner[c][row] = areas[c] / tot;
    }
    __syncthreads();   // gidx visible to all threads

    // Kick off async gather of corner 0's feature rows into sbuf.
    {
        const float* g = feat + gidx[0][grow];
        unsigned l = lds_off_u32(&u.sbuf[grow * 256]);
#pragma unroll
        for (int k = 0; k < 8; ++k)
            async_ld16(l + (unsigned)((gl8 + k * 8) * 16),
                       g + (gl8 + k * 8) * 4);
    }

    v8f o00 = {}, o01 = {}, o10 = {}, o11 = {};

    // ---- 4 corners: staged gather -> MLP (3 WMMA layers) -> accumulate ----
    for (int c = 0; c < 4; ++c) {
        wait_async0();        // my wave's async copies into sbuf are in LDS
        __syncthreads();      // everyone's are; previous readers are done

        // Convert staged f32 rows -> f16 A-tile (+ rel/geo/zero columns)
        {
            const float4* s4 = (const float4*)&u.sbuf[grow * 256];
#pragma unroll
            for (int k = 0; k < 8; ++k) {
                float4 v = s4[gl8 + k * 8];
                _Float16* d = &xbuf[grow * LDX + (gl8 + k * 8) * 4];
                d[0] = (_Float16)v.x; d[1] = (_Float16)v.y;
                d[2] = (_Float16)v.z; d[3] = (_Float16)v.w;
            }
            if (tid < MROWS) {
                int r2 = tid;
                xbuf[r2 * LDX + 256] = (_Float16)relbuf[c][r2][0];
                xbuf[r2 * LDX + 257] = (_Float16)relbuf[c][r2][1];
#pragma unroll
                for (int j = 0; j < 8; ++j)
                    xbuf[r2 * LDX + 258 + j] = (_Float16)geos[r2][j];
#pragma unroll
                for (int j = 266; j < KPAD; ++j)
                    xbuf[r2 * LDX + j] = (_Float16)0.0f;
            }
        }
        __syncthreads();      // xbuf ready; sbuf fully consumed

        // Overlap: async-stage the NEXT corner's rows while we do the GEMMs.
        if (c + 1 < 4) {
            const float* g = feat + gidx[c + 1][grow];
            unsigned l = lds_off_u32(&u.sbuf[grow * 256]);
#pragma unroll
            for (int k = 0; k < 8; ++k)
                async_ld16(l + (unsigned)((gl8 + k * 8) * 16),
                           g + (gl8 + k * 8) * 4);
        }

        // layer 0: (32 x 288) x (288 x 256), ReLU
        v8f a00 = {}, a01 = {}, a10 = {}, a11 = {};
        gemm4(xbuf, LDX, w0f, KPAD / 32, lane, nn, hh, nt0, nt1, a00, a01, a10, a11);
        store_h4(hbuf, LDH, a00, a01, a10, a11, b0, nt0, nt1, hh, nn, true);
        __syncthreads();

        // layer 1: (32 x 256) x (256 x 256), ReLU
        v8f c00 = {}, c01 = {}, c10 = {}, c11 = {};
        gemm4(hbuf, LDH, w1f, HID / 32, lane, nn, hh, nt0, nt1, c00, c01, c10, c11);
        store_h4(hbuf2, LDH, c00, c01, c10, c11, b1, nt0, nt1, hh, nn, true);
        __syncthreads();

        // layer 2: (32 x 256) x (256 x 256) + bias, weighted accumulate
        v8f d00 = {}, d01 = {}, d10 = {}, d11 = {};
        gemm4(hbuf2, LDH, w2f, HID / 32, lane, nn, hh, nt0, nt1, d00, d01, d10, d11);
        float bb0 = b2[nt0 * 16 + nn];
        float bb1 = b2[nt1 * 16 + nn];
#pragma unroll
        for (int r = 0; r < 8; ++r) {
            int row0 = r + 8 * hh;
            int row1 = row0 + 16;
            float w0r = wcorner[c][row0];
            float w1r = wcorner[c][row1];
            o00[r] += w0r * (d00[r] + bb0);
            o01[r] += w0r * (d01[r] + bb1);
            o10[r] += w1r * (d10[r] + bb0);
            o11[r] += w1r * (d11[r] + bb1);
        }
    }

    // ---- final projection: out @ w_out + b_out ----
    __syncthreads();
#pragma unroll
    for (int r = 0; r < 8; ++r) {
        int row0 = r + 8 * hh;
        int row1 = row0 + 16;
        hbuf[row0 * LDH + nt0 * 16 + nn] = (_Float16)o00[r];
        hbuf[row0 * LDH + nt1 * 16 + nn] = (_Float16)o01[r];
        hbuf[row1 * LDH + nt0 * 16 + nn] = (_Float16)o10[r];
        hbuf[row1 * LDH + nt1 * 16 + nn] = (_Float16)o11[r];
    }
    __syncthreads();

    v8f f00 = {}, f01 = {}, f10 = {}, f11 = {};
    gemm4(hbuf, LDH, w3f, HID / 32, lane, nn, hh, nt0, nt1, f00, f01, f10, f11);
    {
        float bb0 = bout[nt0 * 16 + nn];
        float bb1 = bout[nt1 * 16 + nn];
#pragma unroll
        for (int r = 0; r < 8; ++r) {
            int row0 = r + 8 * hh;
            int row1 = row0 + 16;
            u.outf[row0 * LDO + nt0 * 16 + nn] = f00[r] + bb0;
            u.outf[row0 * LDO + nt1 * 16 + nn] = f01[r] + bb1;
            u.outf[row1 * LDO + nt0 * 16 + nn] = f10[r] + bb0;
            u.outf[row1 * LDO + nt1 * 16 + nn] = f11[r] + bb1;
        }
    }
    __syncthreads();

    // ---- LayerNorm over 256 cols per row + write out ----
    {
        float s = 0.0f, s2 = 0.0f;
#pragma unroll
        for (int j = 0; j < 32; ++j) {
            float v = u.outf[grow * LDO + gl8 * 32 + j];
            s += v; s2 += v * v;
        }
        red[0][grow][gl8] = s;
        red[1][grow][gl8] = s2;
    }
    __syncthreads();
    if (tid < MROWS) {
        int row = tid;
        float s = 0.0f, s2 = 0.0f;
#pragma unroll
        for (int j = 0; j < 8; ++j) { s += red[0][row][j]; s2 += red[1][row][j]; }
        float mean = s * (1.0f / 256.0f);
        float var  = s2 * (1.0f / 256.0f) - mean * mean;
        mrow[row] = mean;
        irow[row] = rsqrtf(var + 1e-5f);
    }
    __syncthreads();
    {
        float mean = mrow[grow];
        float inv  = irow[grow];
        size_t obase = ((size_t)(qbase + grow)) * 256;
#pragma unroll
        for (int j = 0; j < 32; ++j) {
            int col = gl8 * 32 + j;
            out[obase + col] =
                (u.outf[grow * LDO + col] - mean) * inv * gamma[col] + beta[col];
        }
    }
}

// ---------------------------------------------------------------------------
// Launch
// ---------------------------------------------------------------------------
extern "C" void kernel_launch(void* const* d_in, const int* in_sizes, int n_in,
                              void* d_out, int out_size, void* d_ws, size_t ws_size,
                              hipStream_t stream)
{
    const float* feat   = (const float*)d_in[0];
    const float* coord  = (const float*)d_in[1];
    const float* geo    = (const float*)d_in[2];
    const float* w0     = (const float*)d_in[3];
    const float* b0     = (const float*)d_in[4];
    const float* w1     = (const float*)d_in[5];
    const float* b1     = (const float*)d_in[6];
    const float* w2     = (const float*)d_in[7];
    const float* b2     = (const float*)d_in[8];
    const float* w_out  = (const float*)d_in[9];
    const float* b_out  = (const float*)d_in[10];
    const float* gamma  = (const float*)d_in[11];
    const float* beta   = (const float*)d_in[12];

    _Float16* ws  = (_Float16*)d_ws;
    _Float16* w0f = ws;                         // 9*16*32*16 = 73728 halves
    _Float16* w1f = w0f + 9 * 16 * 32 * 16;
    _Float16* w2f = w1f + 8 * 16 * 32 * 16;
    _Float16* w3f = w2f + 8 * 16 * 32 * 16;

    pack_w_kernel<<<(9 * 16 * 32 * 16 + 255) / 256, 256, 0, stream>>>(w0,    w0f, 9, INDIM);
    pack_w_kernel<<<(8 * 16 * 32 * 16 + 255) / 256, 256, 0, stream>>>(w1,    w1f, 8, HID);
    pack_w_kernel<<<(8 * 16 * 32 * 16 + 255) / 256, 256, 0, stream>>>(w2,    w2f, 8, HID);
    pack_w_kernel<<<(8 * 16 * 32 * 16 + 255) / 256, 256, 0, stream>>>(w_out, w3f, 8, HID);

    int nblocks = (BATCH * NQ) / MROWS;         // 8192
    svq_main_kernel<<<nblocks, 256, 0, stream>>>(
        feat, coord, geo, b0, b1, b2, b_out, gamma, beta,
        w0f, w1f, w2f, w3f, (float*)d_out);
}
